// FuseView_4827543241002
// MI455X (gfx1250) — matrix operand order repaired and checked by reading, not compiled, not run on hardware
//
#include <hip/hip_runtime.h>
#include <math.h>

typedef _Float16 half_t;
typedef __attribute__((ext_vector_type(16))) _Float16 v16h;
typedef __attribute__((ext_vector_type(8)))  _Float16 v8h;
typedef __attribute__((ext_vector_type(8)))  float    v8f;

#define B_   8
#define T_   128
#define J_   17
#define N_   4
#define C_   544      // J*Cp
#define CIN_ 1088     // 2C
#define CO_  544
#define NPOS 16
#define SH   1096     // padded half-stride per position row (bank-conflict-free)
#define RELS 72       // rel buffer stride (halfs)

// ---------------- prepass kernels (fp32 weights -> f16 in workspace) ----------
__global__ void cvt_f32_to_f16(const float* __restrict__ s, half_t* __restrict__ d, int n) {
  for (int i = blockIdx.x * blockDim.x + threadIdx.x; i < n; i += gridDim.x * blockDim.x)
    d[i] = (half_t)s[i];
}
__global__ void pad_wpos_k(const float* __restrict__ s, half_t* __restrict__ d) {
  int n = CIN_ * 64;
  for (int i = blockIdx.x * blockDim.x + threadIdx.x; i < n; i += gridDim.x * blockDim.x) {
    int o = i >> 6, ch = i & 63;
    d[i] = (ch < 34) ? (half_t)s[o * 34 + ch] : (half_t)0.0f;
  }
}
__global__ void transpose_w(const float* __restrict__ s, half_t* __restrict__ d) {
  int n = CO_ * CO_;
  for (int i = blockIdx.x * blockDim.x + threadIdx.x; i < n; i += gridDim.x * blockDim.x) {
    int dd = i / CO_, c = i % CO_;
    d[i] = (half_t)s[c * CO_ + dd];      // d[dd][c] = src[c][dd]
  }
}

// ---------------- WMMA fragment helpers (per CDNA5 ISA VGPR layouts) ----------
__device__ inline v16h cat8(v8h lo, v8h hi) {
  v16h r;
#pragma unroll
  for (int e = 0; e < 8; ++e) { r[e] = lo[e]; r[e + 8] = hi[e]; }
  return r;
}
// A: 16x32 f16, row-major weights w[o][c], ld halfs per row.
// lane m=lane&15 holds row o0+m; kh=lane>>4 selects K blocks {kh*8..kh*8+7, 16+kh*8..}
__device__ inline v16h loadA(const half_t* __restrict__ w, int ld, int o0, int c0, int lane) {
  int m = lane & 15, kh = lane >> 4;
  const half_t* p = w + (o0 + m) * ld + c0 + kh * 8;
  return cat8(*(const v8h*)p, *(const v8h*)(p + 16));
}
// B: 32x16 f16 from LDS activations buf[pos][channel] (stride halfs).
// lane n=lane&15 is column (position); kh=lane>>4 selects K blocks {kh*8..kh*8+7, 16+kh*8..}
__device__ inline v16h loadB(const half_t* buf, int stride, int c0, int lane) {
  int n = lane & 15, kh = lane >> 4;
  const half_t* p = buf + n * stride + c0 + kh * 8;
  return cat8(*(const v8h*)p, *(const v8h*)(p + 16));
}
__device__ inline v8f wmma16(v16h a, v16h b, v8f c) {
  return __builtin_amdgcn_wmma_f32_16x16x32_f16(false, a, false, b, (short)0, c, false, false);
}
// Fold BN (eval) into per-channel scale/shift in LDS
__device__ inline void bn_st(const float* __restrict__ bn, float* sb, float* tb, int tid) {
  for (int c = tid; c < CIN_; c += 256) {
    float g = bn[c], b = bn[CIN_ + c], m = bn[2 * CIN_ + c], v = bn[3 * CIN_ + c];
    float s = g / sqrtf(v + 1e-5f);
    sb[c] = s; tb[c] = b - m * s;
  }
}
// y = relu(bn(W @ in)) [+ res]  -> out   (1088 out channels, 16 positions)
__device__ inline void gemm_layer(const half_t* __restrict__ w,
                                  const half_t* in, half_t* out,
                                  const float* sb, const float* tb,
                                  const half_t* resb, int wave, int lane) {
  int kh = lane >> 4, n = lane & 15;
  for (int tile = wave; tile < 68; tile += 8) {
    int o0 = tile * 16;
    v8f acc = {};
#pragma unroll 4
    for (int kt = 0; kt < 34; ++kt)
      acc = wmma16(loadA(w, CIN_, o0, kt * 32, lane), loadB(in, SH, kt * 32, lane), acc);
#pragma unroll
    for (int v = 0; v < 8; ++v) {
      int row = o0 + v + 8 * kh;
      float y = fmaxf(acc[v] * sb[row] + tb[row], 0.0f);
      if (resb) y += (float)resb[n * SH + row];
      out[n * SH + row] = (half_t)y;
    }
  }
}

// LDS budget (bytes): floats 17696*4 = 70784 ; halfs 53760*2 = 107520 ; total 178304
#define SMEM_BYTES (70784 + 107520)

extern "C" __global__ void __launch_bounds__(256, 1)
val_fused(const float* __restrict__ x, const float* __restrict__ pos2d,
          const float* __restrict__ b_pos,
          const float* __restrict__ bn_exp, const float* __restrict__ bn1,
          const float* __restrict__ bn2, const float* __restrict__ bn_ag,
          const float* __restrict__ b_ag_s, const float* __restrict__ bn_att,
          const float* __restrict__ w_att_s, const float* __restrict__ b_att_s,
          const half_t* __restrict__ wpos16, const half_t* __restrict__ wexp16,
          const half_t* __restrict__ w1h, const half_t* __restrict__ w2h,
          const half_t* __restrict__ wagh, const half_t* __restrict__ wagsh,
          const half_t* __restrict__ watth,
          const half_t* __restrict__ w0t, const half_t* __restrict__ w1t,
          float* __restrict__ out) {
  extern __shared__ __align__(16) char smem[];
  float*  f32f = (float*)smem;            // 2176  f (fp32, exact residual add)
  float*  fsb  = f32f + 2176;             // 2176  f_self
  float*  ftbf = fsb  + 2176;             // 2176  f_tran
  float*  agb  = ftbf + 2176;             // 8704  ag (544 x 16)
  float*  sb   = agb  + 8704;             // 1088  bn scale
  float*  tb   = sb   + 1088;             // 1088  bn shift
  float*  attb = tb   + 1088;             // 16
  float*  attp = attb + 16;               // 256
  float*  attsm= attp + 256;              // 16
  half_t* b0   = (half_t*)(attsm + 16);   // 16 x SH activation buffers
  half_t* b1   = b0 + NPOS * SH;
  half_t* b2   = b1 + NPOS * SH;
  half_t* relb = b2 + NPOS * SH;          // 16 x 72

  const int tid = threadIdx.x, lane = tid & 31, wave = tid >> 5;
  const int wg = blockIdx.x;                       // b*T + t
  const float* xg = x + (size_t)wg * 2176;         // (c,n): c*4+n
  const float* pg = pos2d + (size_t)wg * 136;      // j*8 + d*4 + n

  // ---- phase 0: f, pairwise h (b0), rel (relb), bn_expand s/t ----
  for (int e = tid; e < 2176; e += 256) f32f[e] = xg[e];
  for (int e = tid; e < NPOS * RELS; e += 256) relb[e] = (half_t)0.0f;
  bn_st(bn_exp, sb, tb, tid);
  __syncthreads();
  for (int e = tid; e < CIN_ * NPOS; e += 256) {
    int c = e >> 4, pos = e & 15, i = pos >> 2, j = pos & 3;
    float v = (c < C_) ? f32f[c * 4 + i] : f32f[(c - C_) * 4 + j];
    b0[pos * SH + c] = (half_t)v;
  }
  for (int e = tid; e < 34 * NPOS; e += 256) {
    int ch = e >> 4, pos = e & 15, i = pos >> 2, j = pos & 3;
    int jj = ch >> 1, d = ch & 1;
    relb[pos * RELS + ch] = (half_t)(pg[jj * 8 + d * 4 + i] - pg[jj * 8 + d * 4 + j]);
  }
  __syncthreads();

  // ---- L0: y0 = relu(bn_exp(Wexp @ h)) + (w_pos @ rel + b_pos)  -> b1 ----
  {
    int kh = lane >> 4, n = lane & 15;
    for (int tile = wave; tile < 68; tile += 8) {
      int o0 = tile * 16;
      v8f acc = {};
#pragma unroll 4
      for (int kt = 0; kt < 34; ++kt)
        acc = wmma16(loadA(wexp16, CIN_, o0, kt * 32, lane), loadB(b0, SH, kt * 32, lane), acc);
      v8f pacc = {};
#pragma unroll
      for (int kt = 0; kt < 2; ++kt)
        pacc = wmma16(loadA(wpos16, 64, o0, kt * 32, lane), loadB(relb, RELS, kt * 32, lane), pacc);
#pragma unroll
      for (int v = 0; v < 8; ++v) {
        int row = o0 + v + 8 * kh;
        float y = fmaxf(acc[v] * sb[row] + tb[row], 0.0f) + pacc[v] + b_pos[row];
        b1[n * SH + row] = (half_t)y;
      }
    }
  }
  __syncthreads();
  bn_st(bn1, sb, tb, tid); __syncthreads();
  gemm_layer(w1h, b1, b2, sb, tb, nullptr, wave, lane);        // y1 -> b2
  __syncthreads();
  bn_st(bn2, sb, tb, tid); __syncthreads();
  gemm_layer(w2h, b2, b0, sb, tb, b1, wave, lane);             // h2 = y0 + y2 -> b0
  __syncthreads();
  bn_st(bn_ag, sb, tb, tid); __syncthreads();
  gemm_layer(wagh, b0, b1, sb, tb, nullptr, wave, lane);       // t_ag -> b1
  __syncthreads();
  bn_st(bn_att, sb, tb, tid);                                  // s/t for att layer
  // ---- AGS: ag = w_ag_s @ t_ag + b_ag_s  (544 x 16, fp32 in LDS) ----
  {
    int kh = lane >> 4, n = lane & 15;
    for (int tile = wave; tile < 34; tile += 8) {
      int o0 = tile * 16;
      v8f acc = {};
#pragma unroll 4
      for (int kt = 0; kt < 34; ++kt)
        acc = wmma16(loadA(wagsh, CIN_, o0, kt * 32, lane), loadB(b1, SH, kt * 32, lane), acc);
#pragma unroll
      for (int v = 0; v < 8; ++v) {
        int row = o0 + v + 8 * kh;
        agb[row * 16 + n] = acc[v] + b_ag_s[row];
      }
    }
  }
  __syncthreads();                                             // b1 free, bn_att visible
  gemm_layer(watth, b0, b2, sb, tb, nullptr, wave, lane);      // t_att -> b2
  for (int e = tid; e < NPOS * SH; e += 256) b1[e] = (half_t)0.0f;   // f buffer
  __syncthreads();
  for (int e = tid; e < C_ * 4; e += 256) {
    int c = e >> 2, n = e & 3;
    b1[n * SH + c] = (half_t)f32f[e];                          // f as WMMA B operand
  }
  // ---- att = w_att_s . t_att + b_att_s (VALU dot, 1 out channel) ----
  {
    int pos = tid >> 4, chunk = tid & 15;
    float s = 0.0f;
    for (int c = chunk * 68; c < chunk * 68 + 68; ++c)
      s += w_att_s[c] * (float)b2[pos * SH + c];
    attp[tid] = s;
  }
  __syncthreads();
  if (tid < 16) {
    float s = 0.0f;
    for (int k = 0; k < 16; ++k) s += attp[tid * 16 + k];
    attb[tid] = s + b_att_s[0];
  }
  __syncthreads();
  if (tid < 4) {                                               // softmax over j per receiver i
    float m = -3.0e38f;
    for (int j = 0; j < 4; ++j) m = fmaxf(m, attb[tid * 4 + j]);
    float ex[4], sum = 0.0f;
    for (int j = 0; j < 4; ++j) { ex[j] = expf(attb[tid * 4 + j] - m); sum += ex[j]; }
    for (int j = 0; j < 4; ++j) attsm[tid * 4 + j] = ex[j] / sum;
  }
  // ---- f_self = W0^T f, f_tran = W1^T f (544 x 4 each; N padded to 16) ----
  {
    int kh = lane >> 4, n = lane & 15;
    for (int tt = wave; tt < 68; tt += 8) {
      const half_t* w = (tt < 34) ? w0t : w1t;
      float* dst = (tt < 34) ? fsb : ftbf;
      int o0 = (tt % 34) * 16;
      v8f acc = {};
#pragma unroll 4
      for (int kt = 0; kt < 17; ++kt)
        acc = wmma16(loadA(w, CO_, o0, kt * 32, lane), loadB(b1, SH, kt * 32, lane), acc);
      if (n < 4) {
#pragma unroll
        for (int v = 0; v < 8; ++v) {
          int row = o0 + v + 8 * kh;
          dst[row * 4 + n] = acc[v];
        }
      }
    }
  }
  __syncthreads();
  // ---- fuse: f_fuse[d,i] = sum_j conv[d,i,j]*att[i,j] + f[d,i] ----
  float* og = out + (size_t)wg * 2176;
  for (int e = tid; e < 2176; e += 256) {
    int d = e >> 2, i = e & 3;
    float sum = 0.0f;
    for (int j = 0; j < 4; ++j) {
      int pos = i * 4 + j;
      float vv = (j == i) ? fsb[d * 4 + j] : ftbf[d * 4 + j] * agb[d * 16 + pos];
      sum += attsm[pos] * vv;
    }
    og[e] = sum + f32f[e];
  }
}

// ------------------------------ host launcher --------------------------------
extern "C" void kernel_launch(void* const* d_in, const int* in_sizes, int n_in,
                              void* d_out, int out_size, void* d_ws, size_t ws_size,
                              hipStream_t stream) {
  const float* x       = (const float*)d_in[0];
  const float* pos2d   = (const float*)d_in[1];
  const float* w_pos   = (const float*)d_in[2];
  const float* b_pos   = (const float*)d_in[3];
  const float* w_exp   = (const float*)d_in[4];
  const float* bn_exp  = (const float*)d_in[5];
  const float* w1      = (const float*)d_in[6];
  const float* bn1     = (const float*)d_in[7];
  const float* w2      = (const float*)d_in[8];
  const float* bn2     = (const float*)d_in[9];
  const float* w_ag    = (const float*)d_in[10];
  const float* bn_ag   = (const float*)d_in[11];
  const float* w_ag_s  = (const float*)d_in[12];
  const float* b_ag_s  = (const float*)d_in[13];
  const float* w_att   = (const float*)d_in[14];
  const float* bn_att  = (const float*)d_in[15];
  const float* w_att_s = (const float*)d_in[16];
  const float* b_att_s = (const float*)d_in[17];
  const float* W       = (const float*)d_in[18];

  // workspace carve (f16 weights, ~14.3 MB; L2-resident on MI455X)
  half_t* ws = (half_t*)d_ws;
  half_t* wpos16 = ws;                       size_t off = (size_t)CIN_ * 64;
  half_t* wexp16 = ws + off;                 off += (size_t)CIN_ * CIN_;
  half_t* w1h    = ws + off;                 off += (size_t)CIN_ * CIN_;
  half_t* w2h    = ws + off;                 off += (size_t)CIN_ * CIN_;
  half_t* wagh   = ws + off;                 off += (size_t)CIN_ * CIN_;
  half_t* wagsh  = ws + off;                 off += (size_t)CO_ * CIN_;
  half_t* watth  = ws + off;                 off += (size_t)CIN_ * CIN_;
  half_t* w0t    = ws + off;                 off += (size_t)CO_ * CO_;
  half_t* w1t    = ws + off;                 off += (size_t)CO_ * CO_;

  // prepass: convert / pad / transpose weights to f16
  cvt_f32_to_f16<<<1024, 256, 0, stream>>>(w_exp,  wexp16, CIN_ * CIN_);
  cvt_f32_to_f16<<<1024, 256, 0, stream>>>(w1,     w1h,    CIN_ * CIN_);
  cvt_f32_to_f16<<<1024, 256, 0, stream>>>(w2,     w2h,    CIN_ * CIN_);
  cvt_f32_to_f16<<<1024, 256, 0, stream>>>(w_ag,   wagh,   CIN_ * CIN_);
  cvt_f32_to_f16<<<1024, 256, 0, stream>>>(w_ag_s, wagsh,  CO_ * CIN_);
  cvt_f32_to_f16<<<1024, 256, 0, stream>>>(w_att,  watth,  CIN_ * CIN_);
  pad_wpos_k   <<<256, 256, 0, stream>>>(w_pos, wpos16);
  transpose_w  <<<512, 256, 0, stream>>>(W,               w0t);
  transpose_w  <<<512, 256, 0, stream>>>(W + CO_ * CO_,   w1t);

  hipFuncSetAttribute((const void*)val_fused,
                      hipFuncAttributeMaxDynamicSharedMemorySize, SMEM_BYTES);
  val_fused<<<B_ * T_, 256, SMEM_BYTES, stream>>>(
      x, pos2d, b_pos, bn_exp, bn1, bn2, bn_ag, b_ag_s, bn_att,
      w_att_s, b_att_s,
      wpos16, wexp16, w1h, w2h, wagh, wagsh, watth, w0t, w1t,
      (float*)d_out);
}
